// HybridSSMAttentionDiT_23175643529605
// MI455X (gfx1250) — matrix operand order repaired and checked by reading, not compiled, not run on hardware
//
#include <hip/hip_runtime.h>
#include <hip/hip_bf16.h>
#include <math.h>

typedef __bf16 bf16;
typedef __attribute__((ext_vector_type(8)))  __bf16 bf16x8;
typedef __attribute__((ext_vector_type(16))) __bf16 v16bf;
typedef __attribute__((ext_vector_type(8)))  float  v8f;

#define NB 4
#define NS 8192
#define NF 512
#define NSTATE 64
#define NH 2048
#define NROWS (NB*NS)        /* 32768 */
#define CHUNK 128
#define NCHUNK (NS/CHUNK)    /* 64 */

// ---------------- workspace layout (bytes) ----------------
static const size_t OFF_ABAR  = 0;                      // 512 f32: abar_re|abar_im|coef_re|coef_im, [2][64] each
static const size_t OFF_MODS  = 4096;                   // 4*3072 f32
static const size_t OFF_E     = 64ull<<10;              // 2*4*64*64 float2 (chunk local ends)
static const size_t OFF_CARRY = 384ull<<10;             // 2*4*64*64 float2 (chunk carries)
static const size_t OFF_BBART = 1ull<<20;               // [256][512] bf16 (B_bar stacked, K-major)
static const size_t OFF_WCT   = (1ull<<20)+(512ull<<10);// [1024][256] bf16 (C weights, K-major)
static const size_t OFF_PROJT = 2ull<<20;               // [512][1024] bf16
static const size_t OFF_W1T   = 3ull<<20;               // [2048][512] bf16
static const size_t OFF_W2T   = 5ull<<20;               // [512][2048] bf16
static const size_t OFF_H     = 16ull<<20;              // [32768][512] bf16
static const size_t OFF_BU    = 48ull<<20;              // [32768][256] f32
static const size_t OFF_XS    = 80ull<<20;              // [32768][256] bf16
static const size_t OFF_YCAT  = 96ull<<20;              // [32768][1024] bf16
static const size_t OFF_T     = 16ull<<20;              // [32768][2048] bf16 (overlays H..YCAT, all dead by then)
static const size_t OFF_X1    = 160ull<<20;             // [32768][512] f32
static const size_t OFF_H2    = 224ull<<20;             // [32768][512] bf16

// ---------------- small math helpers ----------------
__device__ __forceinline__ float silu_f(float x){ return x / (1.0f + __expf(-x)); }
__device__ __forceinline__ float gelu_f(float x){
  float x3 = x*x*x;
  return 0.5f*x*(1.0f + tanhf(0.7978845608028654f*(x + 0.044715f*x3)));
}

// ---------------- WMMA GEMM core (bf16 in, f32 acc) ----------------
// A: [M,K] row-major bf16.  Bt: [N,K] row-major bf16 (i.e. B^T).
// Wave tile 32Mx64N (2x4 wmma frags); block = 8 waves -> 128Mx128N.
__device__ __forceinline__ v16bf ld_frag16(const bf16* p0, const bf16* p1){
  bf16x8 lo = *(const bf16x8*)p0;
  bf16x8 hi = *(const bf16x8*)p1;
  return __builtin_shufflevector(lo, hi, 0,1,2,3,4,5,6,7,8,9,10,11,12,13,14,15);
}

__device__ __forceinline__ void zero_acc(v8f acc[2][4]){
#pragma unroll
  for (int i=0;i<2;i++)
#pragma unroll
    for (int j=0;j<4;j++)
#pragma unroll
      for (int r=0;r<8;r++) acc[i][j][r]=0.0f;
}

__device__ __forceinline__ void gemm_core(const bf16* __restrict__ A, int lda,
                                          const bf16* __restrict__ Bt, int ldb,
                                          int K, int row0, int col0, v8f acc[2][4]){
  const int lane = threadIdx.x & 31;
  const int mi = lane & 15;     // M (or N) index inside 16
  const int kh = lane >> 4;     // K-half selector per ISA 16-bit fragment layout
  for (int kk = 0; kk < K; kk += 32) {
    v16bf a[2];
#pragma unroll
    for (int mt = 0; mt < 2; ++mt) {
      // A 16x32 bf16 fragment: halves 0..7 = K kh*8..kh*8+7, halves 8..15 = K 16+kh*8..
      const bf16* ap = A + (size_t)(row0 + mt*16 + mi)*lda + kk + kh*8;
      a[mt] = ld_frag16(ap, ap + 16);
    }
#pragma unroll
    for (int nt = 0; nt < 4; ++nt) {
      // B 32x16 fragment from K-major B^T: lane column = mi, halves 0..7 = K kh*8..kh*8+7, halves 8..15 = K 16+kh*8..
      const bf16* bp = Bt + (size_t)(col0 + nt*16 + mi)*ldb + kk + kh*8;
      v16bf b = ld_frag16(bp, bp + 16);
#pragma unroll
      for (int mt = 0; mt < 2; ++mt)
        acc[mt][nt] = __builtin_amdgcn_wmma_f32_16x16x32_bf16(
            false, a[mt], false, b, (short)0, acc[mt][nt], false, false);
    }
    if (kk + 32 < K) { // speculative prefetch of next K-slab -> global_prefetch_b8
      __builtin_prefetch(A + (size_t)(row0 + mi)*lda + kk + 32, 0, 1);
      __builtin_prefetch(Bt + (size_t)(col0 + mi)*ldb + kk + 32, 0, 1);
    }
  }
}

#define GEMM_PROLOG(NTOT)                                   \
  const int lane = threadIdx.x & 31;                        \
  const int wave = threadIdx.x >> 5;                        \
  const int row0 = blockIdx.y*128 + (wave & 3)*32;          \
  const int col0 = blockIdx.x*128 + (wave >> 2)*64;         \
  const int kh = lane >> 4; (void)kh;                       \
  v8f acc[2][4]; zero_acc(acc);

// variadic so bodies containing commas survive the preprocessor
#define EPILOGUE_LOOP(...)                                  \
  _Pragma("unroll") for (int mt = 0; mt < 2; ++mt) {        \
    _Pragma("unroll") for (int nt = 0; nt < 4; ++nt) {      \
      const int col = col0 + nt*16 + (lane & 15);           \
      _Pragma("unroll") for (int r = 0; r < 8; ++r) {       \
        const int m = row0 + mt*16 + 8*kh + r;              \
        const float v = acc[mt][nt][r];                     \
        __VA_ARGS__ } } }

// ---------------- prep kernels ----------------
__global__ __launch_bounds__(128) void k_prep_abar(
    const float* lAf, const float* Aif, const float* ldtf,
    const float* lAb, const float* Aib, const float* ldtb, float* abar){
  int t = threadIdx.x; if (t >= 128) return;
  int dir = t >> 6, n = t & 63;
  const float* lA  = dir ? lAb  : lAf;
  const float* Ai  = dir ? Aib  : Aif;
  const float* ldt = dir ? ldtb : ldtf;
  float dt = __expf(ldt[n]);
  float Are = -__expf(lA[n]);
  float Aim = Ai[n];
  float eg = __expf(Are*dt);
  float ab_r = eg*__cosf(Aim*dt), ab_i = eg*__sinf(Aim*dt);
  float Ar2 = Are + 1e-8f;
  float den = Ar2*Ar2 + Aim*Aim;
  float nr = ab_r - 1.0f, ni = ab_i;
  float cr = (nr*Ar2 + ni*Aim)/den;
  float ci = (ni*Ar2 - nr*Aim)/den;
  abar[t] = ab_r; abar[128+t] = ab_i; abar[256+t] = cr; abar[384+t] = ci;
}

// B_bar stacked weight, K-major: rows 0-63 fwd_re, 64-127 fwd_im, 128-191 bwd_re, 192-255 bwd_im
__global__ __launch_bounds__(256) void k_prep_bbart(
    const float* Brf, const float* Bif, const float* Brb, const float* Bib,
    const float* abar, bf16* bbart){
  int idx = blockIdx.x*256 + threadIdx.x;           // 256*512
  int nout = idx >> 9, f = idx & 511;
  int dir = nout >> 7, isim = (nout >> 6) & 1, n = nout & 63;
  const float* Br = dir ? Brb : Brf;
  const float* Bi = dir ? Bib : Bif;
  float cr = abar[256 + dir*64 + n], ci = abar[384 + dir*64 + n];
  float br = Br[n*NF + f], bi = Bi[n*NF + f];
  float v = isim ? (cr*bi + ci*br) : (cr*br - ci*bi);
  bbart[(size_t)nout*NF + f] = (bf16)v;
}

// Wc^T [1024][256]: K rows 0-63 fwd xs_re, 64-127 fwd xs_im, 128-191 bwd re, 192-255 bwd im
__global__ __launch_bounds__(256) void k_prep_wct(
    const float* Crf, const float* Cif, const float* Crb, const float* Cib, bf16* wct){
  int idx = blockIdx.x*256 + threadIdx.x;           // 1024*256
  int n = idx >> 8, k = idx & 255;
  float v = 0.0f;
  if (n < NF) {          // fwd output feature f=n
    if (k < 64)        v =  Crf[n*NSTATE + k];
    else if (k < 128)  v = -Cif[n*NSTATE + (k-64)];
  } else {               // bwd output feature f=n-512
    int f = n - NF;
    if (k >= 128 && k < 192) v =  Crb[f*NSTATE + (k-128)];
    else if (k >= 192)       v = -Cib[f*NSTATE + (k-192)];
  }
  wct[(size_t)n*256 + k] = (bf16)v;
}

__global__ __launch_bounds__(256) void k_prep_projt(const float* pw, bf16* pt){
  int idx = blockIdx.x*256 + threadIdx.x;           // 512*1024
  int n = idx >> 10, k = idx & 1023;
  pt[(size_t)n*1024 + k] = (bf16)pw[(size_t)k*NF + n];
}
__global__ __launch_bounds__(256) void k_prep_w1t(const float* w, bf16* wt){
  int idx = blockIdx.x*256 + threadIdx.x;           // 2048*512
  int n = idx >> 9, k = idx & 511;
  wt[(size_t)n*NF + k] = (bf16)w[(size_t)k*NH + n];
}
__global__ __launch_bounds__(256) void k_prep_w2t(const float* w, bf16* wt){
  int idx = blockIdx.x*256 + threadIdx.x;           // 512*2048
  int n = idx >> 11, k = idx & 2047;
  wt[(size_t)n*NH + k] = (bf16)w[(size_t)k*NF + n];
}

__global__ __launch_bounds__(256) void k_mods(const float* cond, const float* aw,
                                              const float* ab, float* mods){
  int idx = blockIdx.x*256 + threadIdx.x;           // 4*3072
  int b = idx / 3072, j = idx % 3072;
  float s = ab[j];
  for (int i = 0; i < NF; ++i) s += silu_f(cond[b*NF + i]) * aw[(size_t)i*3072 + j];
  mods[idx] = s;
}

// ---------------- LayerNorm + AdaLN modulation ----------------
__global__ __launch_bounds__(256) void k_ln_mod(const float* __restrict__ xin,
                                                const float* __restrict__ mods,
                                                bf16* __restrict__ hout,
                                                int sh_base, int sc_base){
  const int row = blockIdx.x;
  const int b = row >> 13;
  const int t = threadIdx.x;
  float e0 = xin[(size_t)row*NF + t];
  float e1 = xin[(size_t)row*NF + t + 256];
  float s = e0 + e1, q = e0*e0 + e1*e1;
#pragma unroll
  for (int m = 16; m >= 1; m >>= 1) { s += __shfl_xor(s, m, 32); q += __shfl_xor(q, m, 32); }
  __shared__ float red[16];
  const int wave = t >> 5;
  if ((t & 31) == 0) { red[wave] = s; red[8 + wave] = q; }
  __syncthreads();
  float S = 0.f, Q = 0.f;
#pragma unroll
  for (int i = 0; i < 8; ++i) { S += red[i]; Q += red[8+i]; }
  float mu = S * (1.0f/NF);
  float var = Q * (1.0f/NF) - mu*mu;
  float rstd = rsqrtf(var + 1e-5f);
  const float* mb = mods + b*3072;
#pragma unroll
  for (int half = 0; half < 2; ++half) {
    int f = t + half*256;
    float xv = half ? e1 : e0;
    float sc = mb[sc_base + f], sh = mb[sh_base + f];
    hout[(size_t)row*NF + f] = (bf16)(((xv - mu)*rstd)*(1.0f + sc) + sh);
  }
}

// ---------------- GEMM kernels ----------------
// G1: Bu[32768,256] = h @ Bbar^T
__global__ __launch_bounds__(256) void k_gemm_bu(const bf16* __restrict__ h,
                                                 const bf16* __restrict__ bbart,
                                                 float* __restrict__ bu){
  GEMM_PROLOG(256);
  gemm_core(h, NF, bbart, NF, NF, row0, col0, acc);
  EPILOGUE_LOOP( bu[(size_t)m*256 + col] = v; )
}

// G2a: ycat[32768,1024] = xs @ Wc^T + D*h (bf16 out)
__global__ __launch_bounds__(256) void k_gemm_ycat(const bf16* __restrict__ xs,
                                                   const bf16* __restrict__ wct,
                                                   const bf16* __restrict__ h,
                                                   const float* __restrict__ Df,
                                                   const float* __restrict__ Db,
                                                   bf16* __restrict__ ycat){
  GEMM_PROLOG(1024);
  gemm_core(xs, 256, wct, 256, 256, row0, col0, acc);
  EPILOGUE_LOOP(
    float hv;
    float d;
    if (col < NF) { hv = (float)h[(size_t)m*NF + col];        d = Df[col]; }
    else          { hv = (float)h[(size_t)m*NF + (col-NF)];   d = Db[col-NF]; }
    ycat[(size_t)m*1024 + col] = (bf16)(v + d*hv);
  )
}

// G2b: x1[32768,512] = x + g1 * (ycat @ proj^T + proj_b)
__global__ __launch_bounds__(256) void k_gemm_proj(const bf16* __restrict__ ycat,
                                                   const bf16* __restrict__ projt,
                                                   const float* __restrict__ x,
                                                   const float* __restrict__ pb,
                                                   const float* __restrict__ mods,
                                                   float* __restrict__ x1){
  GEMM_PROLOG(512);
  gemm_core(ycat, 1024, projt, 1024, 1024, row0, col0, acc);
  EPILOGUE_LOOP(
    int b = m >> 13;
    float g1 = mods[b*3072 + 2*NF + col];
    x1[(size_t)m*NF + col] = x[(size_t)m*NF + col] + g1*(v + pb[col]);
  )
}

// G3: t[32768,2048] = gelu(h2 @ w1^T + b1) (bf16 out)
__global__ __launch_bounds__(256) void k_gemm_mlp1(const bf16* __restrict__ h2,
                                                   const bf16* __restrict__ w1t,
                                                   const float* __restrict__ b1,
                                                   bf16* __restrict__ tbuf){
  GEMM_PROLOG(2048);
  gemm_core(h2, NF, w1t, NF, NF, row0, col0, acc);
  EPILOGUE_LOOP( tbuf[(size_t)m*NH + col] = (bf16)gelu_f(v + b1[col]); )
}

// G4: out[32768,512] = x1 + g2 * (t @ w2^T + b2)
__global__ __launch_bounds__(256) void k_gemm_mlp2(const bf16* __restrict__ tbuf,
                                                   const bf16* __restrict__ w2t,
                                                   const float* __restrict__ b2,
                                                   const float* __restrict__ x1,
                                                   const float* __restrict__ mods,
                                                   float* __restrict__ out){
  GEMM_PROLOG(512);
  gemm_core(tbuf, NH, w2t, NH, NH, row0, col0, acc);
  EPILOGUE_LOOP(
    int b = m >> 13;
    float g2 = mods[b*3072 + 5*NF + col];
    out[(size_t)m*NF + col] = x1[(size_t)m*NF + col] + g2*(v + b2[col]);
  )
}

// ---------------- chunked complex scan ----------------
// pass1: local ends per chunk.  grid (chunk, b, dir), block 64 (lane = n)
__global__ __launch_bounds__(64) void k_scan_local(const float* __restrict__ bu,
                                                   const float* __restrict__ abar,
                                                   float2* __restrict__ E){
  const int n = threadIdx.x, c = blockIdx.x, b = blockIdx.y, dir = blockIdx.z;
  const float ar = abar[dir*64 + n], ai = abar[128 + dir*64 + n];
  float sr = 0.f, si = 0.f;
  for (int j0 = 0; j0 < CHUNK; ++j0) {
    int j = c*CHUNK + j0;
    int s = dir ? (NS-1 - j) : j;
    size_t row = (size_t)b*NS + s;
    float br = bu[row*256 + dir*128 + n];
    float bi = bu[row*256 + dir*128 + 64 + n];
    float tr = ar*sr - ai*si + br;
    si = ar*si + ai*sr + bi;
    sr = tr;
  }
  E[((size_t)((dir*NB + b)*NSTATE + n))*NCHUNK + c] = make_float2(sr, si);
}

// pass2: sequential carry combine over chunks. grid (b, dir), block 64
__global__ __launch_bounds__(64) void k_scan_carry(const float2* __restrict__ E,
                                                   const float* __restrict__ abar,
                                                   float2* __restrict__ C){
  const int n = threadIdx.x, b = blockIdx.x, dir = blockIdx.y;
  float ar = abar[dir*64 + n], ai = abar[128 + dir*64 + n];
  // A_bar^128 by 7 squarings
  float alr = ar, ali = ai;
#pragma unroll
  for (int i = 0; i < 7; ++i) { float t = alr*alr - ali*ali; ali = 2.f*alr*ali; alr = t; }
  size_t base = ((size_t)((dir*NB + b)*NSTATE + n))*NCHUNK;
  float cr = 0.f, ci = 0.f;
  C[base + 0] = make_float2(0.f, 0.f);
  for (int c = 1; c < NCHUNK; ++c) {
    float2 e = E[base + c - 1];
    float t = alr*cr - ali*ci + e.x;
    ci = alr*ci + ali*cr + e.y;
    cr = t;
    C[base + c] = make_float2(cr, ci);
  }
}

// pass3: replay with carries, emit xs bf16. grid (chunk, b, dir), block 64
__global__ __launch_bounds__(64) void k_scan_emit(const float* __restrict__ bu,
                                                  const float* __restrict__ abar,
                                                  const float2* __restrict__ C,
                                                  bf16* __restrict__ xs){
  const int n = threadIdx.x, c = blockIdx.x, b = blockIdx.y, dir = blockIdx.z;
  const float ar = abar[dir*64 + n], ai = abar[128 + dir*64 + n];
  float2 cin = C[((size_t)((dir*NB + b)*NSTATE + n))*NCHUNK + c];
  float sr = cin.x, si = cin.y;
  for (int j0 = 0; j0 < CHUNK; ++j0) {
    int j = c*CHUNK + j0;
    int s = dir ? (NS-1 - j) : j;
    size_t row = (size_t)b*NS + s;
    float br = bu[row*256 + dir*128 + n];
    float bi = bu[row*256 + dir*128 + 64 + n];
    float tr = ar*sr - ai*si + br;
    si = ar*si + ai*sr + bi;
    sr = tr;
    // bwd results land at row s = flipped position => output already un-flipped
    xs[row*256 + dir*128 + n]      = (bf16)sr;
    xs[row*256 + dir*128 + 64 + n] = (bf16)si;
  }
}

// ---------------- launch ----------------
extern "C" void kernel_launch(void* const* d_in, const int* in_sizes, int n_in,
                              void* d_out, int out_size, void* d_ws, size_t ws_size,
                              hipStream_t stream) {
  (void)in_sizes; (void)n_in; (void)out_size; (void)ws_size;
  const float* x      = (const float*)d_in[0];
  const float* cond   = (const float*)d_in[1];
  const float* f_lA   = (const float*)d_in[2];
  const float* f_Ai   = (const float*)d_in[3];
  const float* f_Br   = (const float*)d_in[4];
  const float* f_Bi   = (const float*)d_in[5];
  const float* f_Cr   = (const float*)d_in[6];
  const float* f_Ci   = (const float*)d_in[7];
  const float* f_D    = (const float*)d_in[8];
  const float* f_ldt  = (const float*)d_in[9];
  const float* b_lA   = (const float*)d_in[10];
  const float* b_Ai   = (const float*)d_in[11];
  const float* b_Br   = (const float*)d_in[12];
  const float* b_Bi   = (const float*)d_in[13];
  const float* b_Cr   = (const float*)d_in[14];
  const float* b_Ci   = (const float*)d_in[15];
  const float* b_D    = (const float*)d_in[16];
  const float* b_ldt  = (const float*)d_in[17];
  const float* proj_w = (const float*)d_in[18];
  const float* proj_b = (const float*)d_in[19];
  const float* ada_w  = (const float*)d_in[20];
  const float* ada_b  = (const float*)d_in[21];
  const float* mlp_w1 = (const float*)d_in[22];
  const float* mlp_b1 = (const float*)d_in[23];
  const float* mlp_w2 = (const float*)d_in[24];
  const float* mlp_b2 = (const float*)d_in[25];
  float* out = (float*)d_out;

  char* ws = (char*)d_ws;
  float*  abar  = (float*) (ws + OFF_ABAR);
  float*  mods  = (float*) (ws + OFF_MODS);
  float2* Ebuf  = (float2*)(ws + OFF_E);
  float2* Cbuf  = (float2*)(ws + OFF_CARRY);
  bf16*   bbart = (bf16*)  (ws + OFF_BBART);
  bf16*   wct   = (bf16*)  (ws + OFF_WCT);
  bf16*   projt = (bf16*)  (ws + OFF_PROJT);
  bf16*   w1t   = (bf16*)  (ws + OFF_W1T);
  bf16*   w2t   = (bf16*)  (ws + OFF_W2T);
  bf16*   hbuf  = (bf16*)  (ws + OFF_H);
  float*  bu    = (float*) (ws + OFF_BU);
  bf16*   xs    = (bf16*)  (ws + OFF_XS);
  bf16*   ycat  = (bf16*)  (ws + OFF_YCAT);
  bf16*   tbuf  = (bf16*)  (ws + OFF_T);
  float*  x1    = (float*) (ws + OFF_X1);
  bf16*   h2    = (bf16*)  (ws + OFF_H2);

  // --- one-time prep (cheap; re-run each call for determinism) ---
  k_prep_abar <<<1, 128, 0, stream>>>(f_lA, f_Ai, f_ldt, b_lA, b_Ai, b_ldt, abar);
  k_prep_bbart<<<512, 256, 0, stream>>>(f_Br, f_Bi, b_Br, b_Bi, abar, bbart);
  k_prep_wct  <<<1024, 256, 0, stream>>>(f_Cr, f_Ci, b_Cr, b_Ci, wct);
  k_prep_projt<<<2048, 256, 0, stream>>>(proj_w, projt);
  k_prep_w1t  <<<4096, 256, 0, stream>>>(mlp_w1, w1t);
  k_prep_w2t  <<<4096, 256, 0, stream>>>(mlp_w2, w2t);
  k_mods      <<<48, 256, 0, stream>>>(cond, ada_w, ada_b, mods);

  // --- main pipeline ---
  k_ln_mod<<<NROWS, 256, 0, stream>>>(x, mods, hbuf, /*sh*/0, /*sc*/NF);

  k_gemm_bu<<<dim3(256/128, NROWS/128), 256, 0, stream>>>(hbuf, bbart, bu);

  k_scan_local<<<dim3(NCHUNK, NB, 2), 64, 0, stream>>>(bu, abar, Ebuf);
  k_scan_carry<<<dim3(NB, 2), 64, 0, stream>>>(Ebuf, abar, Cbuf);
  k_scan_emit <<<dim3(NCHUNK, NB, 2), 64, 0, stream>>>(bu, abar, Cbuf, xs);

  k_gemm_ycat<<<dim3(1024/128, NROWS/128), 256, 0, stream>>>(xs, wct, hbuf, f_D, b_D, ycat);
  k_gemm_proj<<<dim3(512/128, NROWS/128), 256, 0, stream>>>(ycat, projt, x, proj_b, mods, x1);

  k_ln_mod<<<NROWS, 256, 0, stream>>>(x1, mods, h2, /*sh*/3*NF, /*sc*/4*NF);

  k_gemm_mlp1<<<dim3(2048/128, NROWS/128), 256, 0, stream>>>(h2, w1t, mlp_b1, tbuf);
  k_gemm_mlp2<<<dim3(512/128, NROWS/128), 256, 0, stream>>>(tbuf, w2t, mlp_b2, x1, mods, out);
}
